// MixtureOfExperts_82609400971568
// MI455X (gfx1250) — matrix operand order repaired and checked by reading, not compiled, not run on hardware
//
#include <hip/hip_runtime.h>
#include <hip/hip_bf16.h>
#include <math.h>

// Problem constants (match reference)
#define B_  8192
#define D_  1024
#define H_  1024
#define E_  16

// K-chunk staged in LDS and padded LDS row stride (halfs)
#define KC_   128
#define BROW_ (KC_ + 8)   // 136 halfs = 272B row stride -> conflict-spread b128 reads

typedef unsigned short u16;
typedef unsigned int   u32;

typedef __attribute__((ext_vector_type(16))) __bf16 v16bf;
typedef __attribute__((ext_vector_type(8)))  __bf16 v8bf;
typedef __attribute__((ext_vector_type(8)))  float  v8f;

// ---------- helpers ----------
static __device__ __forceinline__ u16 f2bf(float f) {
    u32 u = __float_as_uint(f);
    u = (u + 0x7FFFu + ((u >> 16) & 1u)) >> 16;   // round-to-nearest-even
    return (u16)u;
}

union VU { v16bf v; v8bf h[2]; };
static __device__ __forceinline__ v16bf ld2(const __bf16* p0, const __bf16* p1) {
    VU u;
    u.h[0] = *(const v8bf*)p0;
    u.h[1] = *(const v8bf*)p1;
    return u.v;
}

static __device__ __forceinline__ v8f wmma_bf16(v16bf a, v16bf b, v8f c) {
    return __builtin_amdgcn_wmma_f32_16x16x32_bf16(false, a, false, b, (short)0, c,
                                                   false, false);
}

// ---------- kernel 1: x fp32 -> bf16 ----------
__global__ void k_cvt_x(const float* __restrict__ x, u16* __restrict__ xb) {
    int i = (blockIdx.x * 256 + threadIdx.x) * 4;     // B_*D_ divisible by 1024
    float4 v = *(const float4*)(x + i);
    u32 lo = (u32)f2bf(v.x) | ((u32)f2bf(v.y) << 16);
    u32 hi = (u32)f2bf(v.z) | ((u32)f2bf(v.w) << 16);
    *(uint2*)(xb + i) = make_uint2(lo, hi);
}

// ---------- kernel 2: W1[E,D,H] fp32 -> W1T[E,H,D] bf16 (LDS tile transpose) ----------
__global__ void k_w1t(const float* __restrict__ W1, u16* __restrict__ w1t) {
    __shared__ float t[32][33];
    const int h0 = blockIdx.x * 32, d0 = blockIdx.y * 32, e = blockIdx.z;
    const int tx = threadIdx.x, ty = threadIdx.y;
    #pragma unroll
    for (int r = 0; r < 32; r += 8)
        t[ty + r][tx] = W1[((size_t)e * D_ + (d0 + ty + r)) * H_ + (h0 + tx)]; // t[d][h]
    __syncthreads();
    #pragma unroll
    for (int r = 0; r < 32; r += 8)
        w1t[((size_t)e * H_ + (h0 + ty + r)) * D_ + (d0 + tx)] = f2bf(t[tx][ty + r]);
}

// ---------- kernel 3: gating softmax weights; also zero expert_out ----------
__global__ void k_gate(const float* __restrict__ x, const float* __restrict__ Wg,
                       const float* __restrict__ bg,
                       float* __restrict__ weights, float* __restrict__ eout) {
    const int lane = threadIdx.x & 31;
    const int wv   = threadIdx.x >> 5;
    const int b    = blockIdx.x * 8 + wv;
    const float* xr = x + (size_t)b * D_;
    float acc[E_];
    #pragma unroll
    for (int e = 0; e < E_; ++e) acc[e] = 0.f;
    for (int d = lane; d < D_; d += 32) {
        const float xv = xr[d];
        const float* wr = Wg + (size_t)d * E_;
        #pragma unroll
        for (int e = 0; e < E_; ++e) acc[e] = fmaf(xv, wr[e], acc[e]);
    }
    #pragma unroll
    for (int m = 16; m >= 1; m >>= 1) {
        #pragma unroll
        for (int e = 0; e < E_; ++e) acc[e] += __shfl_xor(acc[e], m, 32);
    }
    if (lane == 0) {
        float lg[E_], mx = -1e30f;
        #pragma unroll
        for (int e = 0; e < E_; ++e) { lg[e] = acc[e] + bg[e]; mx = fmaxf(mx, lg[e]); }
        float s = 0.f;
        #pragma unroll
        for (int e = 0; e < E_; ++e) { lg[e] = __expf(lg[e] - mx); s += lg[e]; }
        const float inv = 1.f / s;
        #pragma unroll
        for (int e = 0; e < E_; ++e) {
            weights[(size_t)b * E_ + e] = lg[e] * inv;
            eout   [(size_t)b * E_ + e] = 0.f;
        }
    }
}

// ---------- kernel 4: fused expert GEMM + bias + ReLU + W2 reduction ----------
// grid = (B/256, H/64, E), block = 256 (8 waves).
// Wave w: rows row0..row0+31 (two 16-row M tiles), 64 cols shared by the block.
// B panel (W1T[e][n0:n0+64][kc:kc+128]) is staged in LDS once per K-chunk and
// read via ds_load_b128 in WMMA fragment layout; A fragments load straight from
// global (L2-resident, per-wave-private rows).
__global__ __launch_bounds__(256, 1)
void k_moe(const u16* __restrict__ xb, const u16* __restrict__ w1t,
           const float* __restrict__ b1, const float* __restrict__ W2,
           float* __restrict__ eout) {
    __shared__ __align__(16) u16 Bp[64 * BROW_];   // ~17.4 KB

    const int tid  = threadIdx.x;
    const int lane = tid & 31;
    const int wv   = tid >> 5;
    const int e    = blockIdx.z;
    const int row0 = blockIdx.x * 256 + wv * 32;
    const int n0   = blockIdx.y * 64;
    const int col  = lane & 15;
    const bool hi  = lane >= 16;

    const __bf16* X  = (const __bf16*)xb;
    const __bf16* Wt = (const __bf16*)w1t;

    // A fragment bases: row = row0+col (+16 for 2nd M tile);
    // lanes 0-15 take K {0..7,16..23}, lanes 16-31 take K {8..15,24..31}
    const __bf16* aptr0 = X + (size_t)(row0 + col) * D_ + (hi ? 8 : 0);
    const __bf16* aptr1 = aptr0 + (size_t)16 * D_;

    // Cooperative B staging: thread t -> n row (t>>2), 64B quarter (t&3)
    const int nl = tid >> 2;      // 0..63
    const int q  = tid & 3;       // 0..3
    const __bf16* bsrc = Wt + ((size_t)e * H_ + (n0 + nl)) * D_ + q * 32;
    u16* bdst = &Bp[nl * BROW_ + q * 32];

    v8f acc0[4] = {}, acc1[4] = {};

    for (int kc = 0; kc < D_; kc += KC_) {
        __syncthreads();                      // previous chunk's reads done
        #pragma unroll
        for (int i = 0; i < 4; ++i)
            *(uint4*)(bdst + i * 8) = *(const uint4*)(bsrc + kc + i * 8);
        __syncthreads();                      // panel visible to all waves

        #pragma unroll
        for (int kk = 0; kk < KC_; kk += 32) {
            const int k0 = kc + kk;
            v16bf a0 = ld2(aptr0 + k0, aptr0 + k0 + 16);
            v16bf a1 = ld2(aptr1 + k0, aptr1 + k0 + 16);
            #pragma unroll
            for (int nt = 0; nt < 4; ++nt) {
                const u16* bp = &Bp[(nt * 16 + col) * BROW_ + kk + (hi ? 8 : 0)];
                v16bf bf = ld2((const __bf16*)bp, (const __bf16*)(bp + 16));
                acc0[nt] = wmma_bf16(a0, bf, acc0[nt]);
                acc1[nt] = wmma_bf16(a1, bf, acc1[nt]);
            }
        }
    }

    // Epilogue: h = relu(acc + b1[e,n]); psum[row] += h * W2[e,n]
    // C/D layout: VGPR j, lanes 0-15 -> row j, lanes 16-31 -> row j+8; col = lane%16.
    float b1v[4], w2v[4];
    #pragma unroll
    for (int nt = 0; nt < 4; ++nt) {
        const int n = n0 + nt * 16 + col;
        b1v[nt] = b1[(size_t)e * H_ + n];
        w2v[nt] = W2[(size_t)e * H_ + n];
    }

    #pragma unroll
    for (int mt = 0; mt < 2; ++mt) {
        float psum[8];
        #pragma unroll
        for (int j = 0; j < 8; ++j) psum[j] = 0.f;
        #pragma unroll
        for (int nt = 0; nt < 4; ++nt) {
            v8f a = mt ? acc1[nt] : acc0[nt];
            #pragma unroll
            for (int j = 0; j < 8; ++j) {
                float h = a[j] + b1v[nt];
                h = fmaxf(h, 0.f);
                psum[j] = fmaf(h, w2v[nt], psum[j]);
            }
        }
        // reduce over the 16 columns held within each half-group of lanes
        #pragma unroll
        for (int m = 8; m >= 1; m >>= 1) {
            #pragma unroll
            for (int j = 0; j < 8; ++j) psum[j] += __shfl_xor(psum[j], m, 32);
        }
        if (col == 0) {
            const int rbase = row0 + mt * 16 + (hi ? 8 : 0);
            #pragma unroll
            for (int j = 0; j < 8; ++j)
                atomicAdd(&eout[(size_t)(rbase + j) * E_ + e], psum[j]);
        }
    }
}

// ---------- kernel 5: weighted combine ----------
__global__ void k_combine(const float* __restrict__ eout, const float* __restrict__ weights,
                          const float* __restrict__ b2, float* __restrict__ out) {
    const int b = blockIdx.x * 256 + threadIdx.x;
    float s = 0.f;
    #pragma unroll
    for (int e = 0; e < E_; ++e)
        s = fmaf(eout[(size_t)b * E_ + e] + b2[e], weights[(size_t)b * E_ + e], s);
    out[b] = s;
}

extern "C" void kernel_launch(void* const* d_in, const int* in_sizes, int n_in,
                              void* d_out, int out_size, void* d_ws, size_t ws_size,
                              hipStream_t stream) {
    const float* x  = (const float*)d_in[0];
    const float* W1 = (const float*)d_in[1];
    const float* b1 = (const float*)d_in[2];
    const float* W2 = (const float*)d_in[3];
    const float* b2 = (const float*)d_in[4];
    const float* Wg = (const float*)d_in[5];
    const float* bg = (const float*)d_in[6];
    float* out = (float*)d_out;

    char* ws = (char*)d_ws;
    u16*   xb      = (u16*)ws;                                   // B*D bf16      = 16 MB
    u16*   w1t     = (u16*)(ws + (size_t)B_ * D_ * 2);           // E*H*D bf16    = 32 MB
    float* weights = (float*)(ws + (size_t)B_ * D_ * 2 + (size_t)E_ * H_ * D_ * 2); // B*E f32
    float* eout    = weights + (size_t)B_ * E_;                  // B*E f32

    // 1. x -> bf16
    k_cvt_x<<<(B_ * D_) / 1024, 256, 0, stream>>>(x, xb);
    // 2. W1 -> transposed bf16 [E,H,D]
    k_w1t<<<dim3(H_ / 32, D_ / 32, E_), dim3(32, 8), 0, stream>>>(W1, w1t);
    // 3. gating softmax (+ zero expert accumulators)
    k_gate<<<B_ / 8, 256, 0, stream>>>(x, Wg, bg, weights, eout);
    // 4. fused expert GEMM
    k_moe<<<dim3(B_ / 256, H_ / 64, E_), 256, 0, stream>>>(xb, w1t, b1, W2, eout);
    // 5. weighted combine
    k_combine<<<B_ / 256, 256, 0, stream>>>(eout, weights, b2, out);
}
